// JanossyPool_6030134083730
// MI455X (gfx1250) — matrix operand order repaired and checked by reading, not compiled, not run on hardware
//
#include <hip/hip_runtime.h>

typedef __attribute__((ext_vector_type(16))) _Float16 v16h;
typedef __attribute__((ext_vector_type(8)))  float    v8f;
typedef __attribute__((ext_vector_type(4)))  float    v4f;

#define B_DIM    64
#define N_SET    32
#define IN_DIMC  128
#define H_DIMC   512
#define OUT_DIMC 256
#define P_PERM   992         // 32*31 permutations of length 2
#define MTILE    64          // P rows per block in stage 2
#define NPT      16          // ceil(992/64)

// ---------------------------------------------------------------------------
// Prep: W2 (f32, [k][g]) -> W2t (f16, [g][k]) so B-fragments are contiguous;
// zero the per-batch hidden-sum accumulator Ssum[B][512].
// ---------------------------------------------------------------------------
__global__ void prep_kernel(const float* __restrict__ W2,
                            _Float16* __restrict__ W2t,
                            float* __restrict__ Ssum) {
    int idx = blockIdx.x * 256 + threadIdx.x;
    if (idx < H_DIMC * H_DIMC) {
        int g = idx / H_DIMC;
        int k = idx % H_DIMC;
        W2t[g * H_DIMC + k] = (_Float16)W2[k * H_DIMC + g];
    }
    if (idx < B_DIM * H_DIMC) Ssum[idx] = 0.0f;
}

// ---------------------------------------------------------------------------
// Stage 1: Apre[b,n,:] = x[b,n,:] @ W1[0:128,:]
//          Bpre[b,n,:] = x[b,n,:] @ W1[128:256,:]
// One block per (b,n) row; 512 threads, one hidden unit each. ~0.5 GFLOP.
// ---------------------------------------------------------------------------
__global__ __launch_bounds__(512)
void stage1_kernel(const float* __restrict__ x, const float* __restrict__ W1,
                   float* __restrict__ Apre, float* __restrict__ Bpre) {
    __shared__ float xs[IN_DIMC];
    const int bn  = blockIdx.x;       // 0..2047
    const int tid = threadIdx.x;      // 0..511
    if (tid < IN_DIMC) xs[tid] = x[bn * IN_DIMC + tid];
    __syncthreads();
    float a = 0.0f, bb = 0.0f;
    #pragma unroll 8
    for (int d = 0; d < IN_DIMC; ++d) {
        float xv = xs[d];
        a  += xv * W1[d * H_DIMC + tid];
        bb += xv * W1[(IN_DIMC + d) * H_DIMC + tid];
    }
    Apre[bn * H_DIMC + tid] = a;
    Bpre[bn * H_DIMC + tid] = bb;
}

// ---------------------------------------------------------------------------
// Stage 2 (heavy, 33 GFLOP): per (b, 64-row P tile):
//   h1 = relu(Apre[i] + Bpre[j] + b1)  (f16, staged in LDS)
//   h2 = relu(h1 @ W2 + b2)            (WMMA f32_16x16x32_f16)
//   Ssum[b,:] += sum over valid rows of h2   (masked for P padding)
//
// Tiling: 16 waves; wave w owns column tiles {w, w+16} for ALL 4 row slabs
// (8 accumulators). Each W2 fragment is loaded exactly once per block and
// reused by 4 WMMAs; B fragments are double-buffered so the next k-step's
// global loads overlap the current 8-WMMA chain.
// ---------------------------------------------------------------------------
__global__ __launch_bounds__(512)
void stage2_kernel(const float* __restrict__ Apre, const float* __restrict__ Bpre,
                   const float* __restrict__ b1,
                   const _Float16* __restrict__ W2t, const float* __restrict__ b2,
                   float* __restrict__ Ssum) {
    __shared__ _Float16 h1s[MTILE][H_DIMC];   // 64 KB
    __shared__ float    Spart[H_DIMC];        // 2 KB

    const int b   = blockIdx.y;
    const int p0  = blockIdx.x * MTILE;
    const int tid = threadIdx.x;

    if (tid < H_DIMC) Spart[tid] = 0.0f;

    // ---- stage h1 tile into LDS (coalesced: consecutive tid -> consecutive c)
    #pragma unroll 4
    for (int it = 0; it < (MTILE * H_DIMC) / 512; ++it) {
        int idx = it * 512 + tid;
        int r = idx >> 9;          // / 512
        int c = idx & (H_DIMC - 1);
        int p = p0 + r;
        float v = 0.0f;
        if (p < P_PERM) {
            int i  = p / (N_SET - 1);
            int jj = p % (N_SET - 1);
            int j  = jj + (jj >= i ? 1 : 0);
            v = Apre[(b * N_SET + i) * H_DIMC + c]
              + Bpre[(b * N_SET + j) * H_DIMC + c] + b1[c];
            v = v > 0.0f ? v : 0.0f;
        }
        h1s[r][c] = (_Float16)v;
    }
    __syncthreads();

    const int w    = tid >> 5;     // wave 0..15
    const int lane = tid & 31;
    const int lm   = lane & 15;
    const int half = lane >> 4;

    union Frag { v16h h; v4f f[2]; };

    v8f zero = {};
    v8f acc[2][4];                 // [ctile 0/1][row slab 0..3]
    #pragma unroll
    for (int c = 0; c < 2; ++c)
        #pragma unroll
        for (int m = 0; m < 4; ++m) acc[c][m] = zero;

    // Per-lane B base pointers: columns n0 = w*16+lm, n1 = (w+16)*16+lm.
    // B fragment layout: VGPR j holds K = 2j + 16*half -> 32 contiguous bytes.
    const _Float16* bbase0 = W2t + (w * 16 + lm) * H_DIMC + 16 * half;
    const _Float16* bbase1 = W2t + ((w + 16) * 16 + lm) * H_DIMC + 16 * half;

    Frag bf[2][2];                 // double buffer x 2 column tiles
    bf[0][0].f[0] = *(const v4f*)(bbase0);
    bf[0][0].f[1] = *(const v4f*)(bbase0 + 8);
    bf[0][1].f[0] = *(const v4f*)(bbase1);
    bf[0][1].f[1] = *(const v4f*)(bbase1 + 8);

    for (int ks = 0; ks < H_DIMC / 32; ++ks) {
        const int cur = ks & 1;
        const int nxt = cur ^ 1;
        const int k0  = ks * 32;

        // A fragments for the 4 row slabs (contiguous k order used here:
        // k = {0..15}+16*half, matching B).
        Frag a[4];
        #pragma unroll
        for (int m = 0; m < 4; ++m) {
            a[m].f[0] = *(const v4f*)&h1s[m * 16 + lm][k0 + 16 * half];
            a[m].f[1] = *(const v4f*)&h1s[m * 16 + lm][k0 + 16 * half + 8];
        }

        // Prefetch next k-step's B fragments (wrapped index keeps the code
        // branch-free; the final prefetch is harmlessly discarded).
        const int kn = ((ks + 1) & (H_DIMC / 32 - 1)) * 32;
        bf[nxt][0].f[0] = *(const v4f*)(bbase0 + kn);
        bf[nxt][0].f[1] = *(const v4f*)(bbase0 + kn + 8);
        bf[nxt][1].f[0] = *(const v4f*)(bbase1 + kn);
        bf[nxt][1].f[1] = *(const v4f*)(bbase1 + kn + 8);

        // 8 WMMAs: each B fragment reused by 4 row slabs.
        #pragma unroll
        for (int c = 0; c < 2; ++c)
            #pragma unroll
            for (int m = 0; m < 4; ++m)
                acc[c][m] = __builtin_amdgcn_wmma_f32_16x16x32_f16(
                    false, a[m].h, false, bf[cur][c].h, (short)0,
                    acc[c][m], false, false);
    }

    // ---- bias + relu + permutation-masked row sum -> LDS -> global atomic
    #pragma unroll
    for (int c = 0; c < 2; ++c) {
        const int n = (w + 16 * c) * 16 + lm;
        const float bias = b2[n];
        float s = 0.0f;
        #pragma unroll
        for (int m = 0; m < 4; ++m) {
            #pragma unroll
            for (int v = 0; v < 8; ++v) {
                // C/D layout: VGPR v holds row m = v + 8*half
                const int prow = p0 + m * 16 + 8 * half + v;
                float h2 = acc[c][m][v] + bias;
                h2 = h2 > 0.0f ? h2 : 0.0f;
                if (prow < P_PERM) s += h2;
            }
        }
        atomicAdd(&Spart[n], s);   // ds_add_f32 (two lanes share each n)
    }
    __syncthreads();
    if (tid < H_DIMC) atomicAdd(&Ssum[b * H_DIMC + tid], Spart[tid]);
}

// ---------------------------------------------------------------------------
// Stage 3: out[b,:] = Ssum[b,:] @ W3 + P * b3   (tiny: 17 MFLOP, plain VALU)
// ---------------------------------------------------------------------------
__global__ __launch_bounds__(256)
void stage3_kernel(const float* __restrict__ Ssum, const float* __restrict__ W3,
                   const float* __restrict__ b3, float* __restrict__ out) {
    __shared__ float s[H_DIMC];
    const int b = blockIdx.x;
    const int o = threadIdx.x;   // 0..255
    for (int g = o; g < H_DIMC; g += OUT_DIMC) s[g] = Ssum[b * H_DIMC + g];
    __syncthreads();
    float acc = (float)P_PERM * b3[o];
    #pragma unroll 8
    for (int g = 0; g < H_DIMC; ++g) acc += s[g] * W3[g * OUT_DIMC + o];
    out[b * OUT_DIMC + o] = acc;
}

// ---------------------------------------------------------------------------
extern "C" void kernel_launch(void* const* d_in, const int* in_sizes, int n_in,
                              void* d_out, int out_size, void* d_ws, size_t ws_size,
                              hipStream_t stream) {
    const float* x  = (const float*)d_in[0];
    const float* W1 = (const float*)d_in[1];
    const float* b1 = (const float*)d_in[2];
    const float* W2 = (const float*)d_in[3];
    const float* b2 = (const float*)d_in[4];
    const float* W3 = (const float*)d_in[5];
    const float* b3 = (const float*)d_in[6];

    char* ws = (char*)d_ws;
    float*    Apre = (float*)(ws);                         // 2048*512*4 = 4 MB
    float*    Bpre = (float*)(ws + 4194304);               // 4 MB
    float*    Ssum = (float*)(ws + 8388608);               // 64*512*4 = 128 KB
    _Float16* W2t  = (_Float16*)(ws + 8519680);            // 512*512*2 = 512 KB

    prep_kernel<<<(H_DIMC * H_DIMC + 255) / 256, 256, 0, stream>>>(W2, W2t, Ssum);
    stage1_kernel<<<B_DIM * N_SET, H_DIMC, 0, stream>>>(x, W1, Apre, Bpre);
    dim3 g2(NPT, B_DIM);
    stage2_kernel<<<g2, 512, 0, stream>>>(Apre, Bpre, b1, W2t, b2, Ssum);
    stage3_kernel<<<B_DIM, OUT_DIMC, 0, stream>>>(Ssum, W3, b3, (float*)d_out);
}
